// SINCERELoss_32684701123302
// MI455X (gfx1250) — matrix operand order repaired and checked by reading, not compiled, not run on hardware
//
#include <hip/hip_runtime.h>
#include <hip/hip_bf16.h>

// SINCERE loss, fused, WMMA-based. B=8192, D=512, 100 classes, T=0.07.

#define BN   8192
#define DK   512
#define NCLS 100

typedef _Float16 v16h __attribute__((ext_vector_type(16)));
typedef _Float16 v8h  __attribute__((ext_vector_type(8)));
typedef _Float16 v4h  __attribute__((ext_vector_type(4)));
typedef float    v8f  __attribute__((ext_vector_type(8)));

union ABFrag { v16h v; v8h h[2]; };

__global__ void sincere_zero(int* cnt, float* out) {
    int t = threadIdx.x;
    if (t < NCLS) cnt[t] = 0;
    if (t == 0)   out[0] = 0.0f;
}

// fp32 -> fp16 convert of embeds (4 elems / thread)
__global__ void sincere_cvt(const float* __restrict__ in, _Float16* __restrict__ out) {
    int i = (blockIdx.x * blockDim.x + threadIdx.x) * 4;
    float4 f = *(const float4*)(in + i);
    v4h h;
    h[0] = (_Float16)f.x; h[1] = (_Float16)f.y;
    h[2] = (_Float16)f.z; h[3] = (_Float16)f.w;
    *(v4h*)(out + i) = h;
}

__global__ void sincere_hist(const int* __restrict__ labels, int* cnt) {
    int i = blockIdx.x * blockDim.x + threadIdx.x;
    if (i < BN) atomicAdd(&cnt[labels[i]], 1);
}

__global__ __launch_bounds__(256, 1)
void sincere_main(const _Float16* __restrict__ e16,
                  const int* __restrict__ labels,
                  const int* __restrict__ cnt,
                  float* __restrict__ out) {
    __shared__ float s_row[64];   // per-row sum(exp) over different-label cols, then log of it
    __shared__ float s_loss;

    constexpr float INV_T = 1.0f / 0.07f;

    const int tid  = threadIdx.x;
    const int w    = tid >> 5;        // wave 0..7
    const int lane = tid & 31;
    const int h    = lane >> 4;       // half-wave
    const int l16  = lane & 15;
    const int mw   = blockIdx.x * 64 + (w & 3) * 16;   // wave's 16-row tile base
    const int nbase = (w >> 2) * 16;                   // column phase (0 or 16)

    if (tid < 64) s_row[tid] = 0.0f;
    if (tid == 0) s_loss = 0.0f;
    __syncthreads();

    // ---- A fragments for this wave's 16 rows, full K=512, kept in VGPRs ----
    // CDNA5 16-bit A 16x32 layout: lane<16 : M=l16, K={k0..k0+7, k0+16..k0+23}
    //                              lane>=16: M=l16, K={k0+8..k0+15, k0+24..k0+31}
    const _Float16* arow = e16 + (size_t)(mw + l16) * DK;
    ABFrag a[16];
#pragma unroll
    for (int kt = 0; kt < 16; ++kt) {
        const int k0 = kt * 32;
        a[kt].h[0] = *(const v8h*)(arow + k0 + h * 8);
        a[kt].h[1] = *(const v8h*)(arow + k0 + 16 + h * 8);
    }

    // C/D layout: lane holds column N=l16, VGPR r holds row M = myrow0 + r
    const int myrow0 = mw + 8 * h;
    int labrow[8];
#pragma unroll
    for (int r = 0; r < 8; ++r) labrow[r] = labels[myrow0 + r];

    // ================= pass 1: denominator sum(exp) =================
    float sacc[8];
#pragma unroll
    for (int r = 0; r < 8; ++r) sacc[r] = 0.0f;

    for (int nt = 0; nt < BN / 32; ++nt) {
        const int n0 = nt * 32 + nbase;
        const _Float16* bcol = e16 + (size_t)(n0 + l16) * DK;  // B[k,n] = E[n,k]
        v8f c = {};
#pragma unroll
        for (int kt = 0; kt < 16; ++kt) {
            // B 32x16 layout: lane<16: N=l16, K={0..7,16..23} of k-tile; lane>=16: K={8..15,24..31}
            ABFrag b;
            b.h[0] = *(const v8h*)(bcol + kt * 32 + 8 * h);
            b.h[1] = *(const v8h*)(bcol + kt * 32 + 16 + 8 * h);
            c = __builtin_amdgcn_wmma_f32_16x16x32_f16(
                    false, a[kt].v, false, b.v, (short)0, c, false, false);
        }
        const int labcol = labels[n0 + l16];
        // |dot| <= 1 (unit vectors) -> exp(logit) <= e^14.3, fp32-safe, no max trick
#pragma unroll
        for (int r = 0; r < 8; ++r) {
            if (labrow[r] != labcol) sacc[r] += __expf(c[r] * INV_T);
        }
    }
#pragma unroll
    for (int r = 0; r < 8; ++r)
        atomicAdd(&s_row[(w & 3) * 16 + 8 * h + r], sacc[r]);
    __syncthreads();
    if (tid < 64) s_row[tid] = __logf(s_row[tid]);
    __syncthreads();

    float neglse[8], invc[8];
#pragma unroll
    for (int r = 0; r < 8; ++r) {
        neglse[r] = s_row[(w & 3) * 16 + 8 * h + r];
        const int cc = cnt[labrow[r]] - 1;
        invc[r] = (cc > 0) ? (1.0f / (float)cc) : 0.0f;
    }

    // ================= pass 2: numerator terms =================
    float lacc = 0.0f;
    for (int nt = 0; nt < BN / 32; ++nt) {
        const int n0 = nt * 32 + nbase;
        const _Float16* bcol = e16 + (size_t)(n0 + l16) * DK;
        v8f c = {};
#pragma unroll
        for (int kt = 0; kt < 16; ++kt) {
            ABFrag b;
            b.h[0] = *(const v8h*)(bcol + kt * 32 + 8 * h);
            b.h[1] = *(const v8h*)(bcol + kt * 32 + 16 + 8 * h);
            c = __builtin_amdgcn_wmma_f32_16x16x32_f16(
                    false, a[kt].v, false, b.v, (short)0, c, false, false);
        }
        const int col = n0 + l16;
        const int labcol = labels[col];
#pragma unroll
        for (int r = 0; r < 8; ++r) {
            if (labrow[r] == labcol && (myrow0 + r) != col) {
                // logaddexp(l, lse) - l = log1p(exp(lse - l))
                const float lg = c[r] * INV_T;
                lacc += invc[r] * log1pf(__expf(neglse[r] - lg));
            }
        }
    }
    atomicAdd(&s_loss, lacc);
    __syncthreads();
    if (tid == 0) atomicAdd(out, s_loss * (1.0f / (float)BN));
}

extern "C" void kernel_launch(void* const* d_in, const int* in_sizes, int n_in,
                              void* d_out, int out_size, void* d_ws, size_t ws_size,
                              hipStream_t stream) {
    (void)in_sizes; (void)n_in; (void)out_size; (void)ws_size;
    const float* embeds = (const float*)d_in[0];
    const int*   labels = (const int*)d_in[1];
    float*       out    = (float*)d_out;

    // workspace layout: [0, 8MB) fp16 embeds ; then 100 ints of class counts
    _Float16* e16 = (_Float16*)d_ws;
    int*      cnt = (int*)((char*)d_ws + (size_t)BN * DK * sizeof(_Float16));

    sincere_zero<<<1, 128, 0, stream>>>(cnt, out);
    sincere_cvt<<<(BN * DK / 4) / 256, 256, 0, stream>>>(embeds, e16);
    sincere_hist<<<BN / 256, 256, 0, stream>>>(labels, cnt);
    sincere_main<<<BN / 64, 256, 0, stream>>>(e16, labels, cnt, out);
}
